// SemiCRF_403726926303
// MI455X (gfx1250) — matrix operand order repaired and checked by reading, not compiled, not run on hardware
//
#include <hip/hip_runtime.h>
#include <hip/hip_fp16.h>

typedef __attribute__((ext_vector_type(16))) _Float16 v16h;
typedef __attribute__((ext_vector_type(8)))  float    v8f;

#define S_LEN 1024
#define BATCH 128
#define NTAG  32
#define SEGL  8
#define NEGV  -10000.0f

__device__ __forceinline__ float half_max16(float v) {
    // butterfly within each 16-lane half (masks 1,2,4,8 stay in-half)
    #pragma unroll
    for (int m = 1; m <= 8; m <<= 1) v = fmaxf(v, __shfl_xor(v, m, 32));
    return v;
}
__device__ __forceinline__ float half_sum16(float v) {
    #pragma unroll
    for (int m = 1; m <= 8; m <<= 1) v += __shfl_xor(v, m, 32);
    return v;
}
__device__ __forceinline__ float wave_sum32(float v) {
    #pragma unroll
    for (int m = 1; m <= 16; m <<= 1) v += __shfl_xor(v, m, 32);
    return v;
}

// ---------------------------------------------------------------------------
// DP / partition-function kernel: one wave handles TWO batches.
// A-matrix rows = (batch-in-pair, i) = 16 rows, K = 32 (prev-tag or emission dim).
// ---------------------------------------------------------------------------
__global__ void __launch_bounds__(32) semicrf_dp(
    const float* __restrict__ em,      // (S,B,T)
    const float* __restrict__ start_t, // (T)
    const float* __restrict__ end_t,   // (T)
    const float* __restrict__ trans,   // (T,T)
    const float* __restrict__ wpool,   // (T,T)
    const float* __restrict__ bpool,   // (T)
    float* __restrict__ denom_out)     // (B)
{
    __shared__ float ring[9][2][NTAG];     // csum ring, slots q mod 9
    __shared__ float wbuf[2][SEGL][NTAG];  // exp-weight transpose staging

    const int lane = threadIdx.x;
    const int b0   = blockIdx.x * 2;

    // ----- D-layout role: (bb, t) ownership -----
    const int bb  = lane >> 4;       // 0 -> batch b0, 1 -> batch b0+1
    const int tl  = lane & 15;
    const int tA  = tl, tB = tl + 16;
    const int myb = b0 + bb;

    // ----- A-layout role: row M = lane&15 -> (abb, aii); K half by lane>>4 -----
    const int rowM  = lane & 15;
    const int abb   = rowM >> 3;
    const int aii   = rowM & 7;
    const int kbaseA = (lane >> 4) * 8;    // unused; A operands use kbaseB, K in {0..15} or {16..31}
    const int kbaseB = (lane >> 4) * 16;   // B: K in {0..15} or {16..31}
    const int ncol   = lane & 15;

    // init csum ring to zero
    #pragma unroll
    for (int q = 0; q < 9; ++q) { ring[q][bb][tA] = 0.f; ring[q][bb][tB] = 0.f; }

    // ----- one-time B operands: E = exp(trans) and Wp (pool), WMMA-B layout -----
    v16h E0, E1, W0, W1;
    #pragma unroll
    for (int e = 0; e < 16; ++e) {
        int kA = kbaseA + (e < 8 ? e : e + 8);   // (unused marker to silence nothing)
        (void)kA;
        int k = kbaseB + e;
        E0[e] = (_Float16)__expf(trans[k * NTAG + ncol]);
        E1[e] = (_Float16)__expf(trans[k * NTAG + (ncol + 16)]);
        W0[e] = (_Float16)wpool[ncol * NTAG + k];
        W1[e] = (_Float16)wpool[(ncol + 16) * NTAG + k];
    }

    const float bpA = bpool[tA], bpB = bpool[tB];

    // ----- j = 0: emission csum + alpha0 -----
    float csA = em[(0 * BATCH + myb) * NTAG + tA];
    float csB = em[(0 * BATCH + myb) * NTAG + tB];
    ring[0][bb][tA] = csA; ring[0][bb][tB] = csB;
    asm volatile("s_wait_dscnt 0" ::: "memory");

    float a0A = start_t[tA] + bpA, a0B = start_t[tB] + bpB;
    #pragma unroll 4
    for (int p = 0; p < NTAG; ++p) {
        float ep = ring[0][bb][p];
        a0A += ep * wpool[tA * NTAG + p];
        a0B += ep * wpool[tB * NTAG + p];
    }

    float bufA[SEGL], bufB[SEGL];   // alpha ring: buf[i] = alpha_{j-1-i}
    bufA[0] = a0A; bufB[0] = a0B;
    #pragma unroll
    for (int i = 1; i < SEGL; ++i) { bufA[i] = NEGV; bufB[i] = NEGV; }

    #pragma unroll 1
    for (int j = 1; j < S_LEN; ++j) {
        // phase 0: streaming csum update (L2-resident emissions)
        csA += em[(j * BATCH + myb) * NTAG + tA];
        csB += em[(j * BATCH + myb) * NTAG + tB];
        const int slj = j % 9;
        ring[slj][bb][tA] = csA; ring[slj][bb][tB] = csB;

        // phase 1: per-(i,b) running max + exp-weights, staged to LDS for transpose
        float mreg[SEGL];
        #pragma unroll
        for (int i = 0; i < SEGL; ++i) {
            float m = half_max16(fmaxf(bufA[i], bufB[i]));
            mreg[i] = m;
            wbuf[bb][i][tA] = __expf(bufA[i] - m);
            wbuf[bb][i][tB] = __expf(bufB[i] - m);
        }
        asm volatile("s_wait_dscnt 0" ::: "memory");

        // phase 2: gather A operands, row M per lane (contiguous K halves, same k order as B)
        const int qp  = j - 1 - aii;           // >= -7
        const int slp = (qp + 9) % 9;          // unwritten slots are still zero
        v16h Aw, As;
        #pragma unroll
        for (int e = 0; e < 16; ++e) {
            int k = kbaseB + e;
            Aw[e] = (_Float16)wbuf[abb][aii][k];
            As[e] = (_Float16)(ring[slj][abb][k] - ring[slp][abb][k]);
        }

        // phase 3: 4x WMMA  (D rows = (batch,i); lanes already hold their own batch)
        v8f z0 = {}, z1 = {}, z2 = {}, z3 = {};
        v8f Dt0 = __builtin_amdgcn_wmma_f32_16x16x32_f16(false, Aw, false, E0, (short)0, z0, false, false);
        v8f Dt1 = __builtin_amdgcn_wmma_f32_16x16x32_f16(false, Aw, false, E1, (short)0, z1, false, false);
        v8f Dp0 = __builtin_amdgcn_wmma_f32_16x16x32_f16(false, As, false, W0, (short)0, z2, false, false);
        v8f Dp1 = __builtin_amdgcn_wmma_f32_16x16x32_f16(false, As, false, W1, (short)0, z3, false, false);

        // phase 4: alpha_j[t] = log sum_i Dt_i * exp(m_i + epool_i - M)   (+M)
        const int ilim = (j < SEGL) ? j : SEGL;
        float gA[SEGL], gB[SEGL];
        float M2A = NEGV, M2B = NEGV;
        #pragma unroll
        for (int i = 0; i < SEGL; ++i) {
            gA[i] = mreg[i] + Dp0[i] + bpA;
            gB[i] = mreg[i] + Dp1[i] + bpB;
            if (i < ilim) { M2A = fmaxf(M2A, gA[i]); M2B = fmaxf(M2B, gB[i]); }
        }
        float sA = 0.f, sB = 0.f;
        #pragma unroll
        for (int i = 0; i < SEGL; ++i) {
            float ea = Dt0[i] * __expf(gA[i] - M2A);
            float eb = Dt1[i] * __expf(gB[i] - M2B);
            if (i < ilim) { sA += ea; sB += eb; }
        }
        float alA = M2A + __logf(sA);
        float alB = M2B + __logf(sB);

        // phase 5: shift alpha ring
        #pragma unroll
        for (int i = SEGL - 1; i > 0; --i) { bufA[i] = bufA[i - 1]; bufB[i] = bufB[i - 1]; }
        bufA[0] = alA; bufB[0] = alB;
    }

    // denom_b = logsumexp_t(alpha_last + end_transitions)
    float vA = bufA[0] + end_t[tA], vB = bufB[0] + end_t[tB];
    float mm = half_max16(fmaxf(vA, vB));
    float ss = half_sum16(__expf(vA - mm) + __expf(vB - mm));
    if (tl == 0) denom_out[myb] = mm + __logf(ss);
}

// ---------------------------------------------------------------------------
// Gold-path score: one wave per batch, lane = emission dim.
// ---------------------------------------------------------------------------
__global__ void __launch_bounds__(32) semicrf_score(
    const float* __restrict__ em, const int* __restrict__ tags,
    const float* __restrict__ start_t, const float* __restrict__ end_t,
    const float* __restrict__ trans, const float* __restrict__ wpool,
    const float* __restrict__ bpool, float* __restrict__ score_out)
{
    const int b = blockIdx.x;
    const int p = threadIdx.x;
    __shared__ float wp_s[NTAG * NTAG];
    #pragma unroll 4
    for (int r = 0; r < NTAG; ++r) wp_s[r * NTAG + p] = wpool[r * NTAG + p];
    __syncthreads();

    int   cur    = tags[0 * BATCH + b];
    int   run    = 1;
    float segsum = em[(0 * BATCH + b) * NTAG + p];
    float acc    = 0.f;               // per-lane dot partials
    float sca    = start_t[cur];      // lane-uniform scalar terms

    #pragma unroll 1
    for (int j = 1; j < S_LEN; ++j) {
        int   t = tags[j * BATCH + b];
        float e = em[(j * BATCH + b) * NTAG + p];
        bool  nb = (t != cur) || (run == SEGL);
        if (nb) {
            acc += segsum * wp_s[cur * NTAG + p];        // close segment: pooled dot
            sca += bpool[cur] + trans[cur * NTAG + t];   // bias + transition
            segsum = e; run = 1; cur = t;
        } else {
            segsum += e; run++;
        }
    }
    acc += segsum * wp_s[cur * NTAG + p];
    sca += bpool[cur] + end_t[cur];

    float tot = wave_sum32(acc);
    if (p == 0) score_out[b] = tot + sca;
}

// ---------------------------------------------------------------------------
// Deterministic final reduction: out = sum_b (score_b - denom_b)
// ---------------------------------------------------------------------------
__global__ void __launch_bounds__(32) semicrf_reduce(
    const float* __restrict__ score, const float* __restrict__ denom,
    float* __restrict__ out)
{
    float acc = 0.f;
    #pragma unroll 4
    for (int b = threadIdx.x; b < BATCH; b += 32) acc += score[b] - denom[b];
    acc = wave_sum32(acc);
    if (threadIdx.x == 0) out[0] = acc;
}

extern "C" void kernel_launch(void* const* d_in, const int* in_sizes, int n_in,
                              void* d_out, int out_size, void* d_ws, size_t ws_size,
                              hipStream_t stream) {
    const float* em      = (const float*)d_in[0];
    const int*   tags    = (const int*)  d_in[1];
    // d_in[2] = mask (all ones in this problem; semantics assumed full-length)
    const float* start_t = (const float*)d_in[3];
    const float* end_t   = (const float*)d_in[4];
    const float* trans   = (const float*)d_in[5];
    const float* wpool   = (const float*)d_in[6];
    const float* bpool   = (const float*)d_in[7];

    float* ws    = (float*)d_ws;
    float* score = ws;            // 128 floats
    float* denom = ws + BATCH;    // 128 floats

    semicrf_score<<<BATCH, 32, 0, stream>>>(em, tags, start_t, end_t, trans, wpool, bpool, score);
    semicrf_dp<<<BATCH / 2, 32, 0, stream>>>(em, start_t, end_t, trans, wpool, bpool, denom);
    semicrf_reduce<<<1, 32, 0, stream>>>(score, denom, (float*)d_out);
}
